// AdjGenerator_68728066670624
// MI455X (gfx1250) — compile-verified, matrix operand order checked by reading, not run on hardware
//
#include <hip/hip_runtime.h>
#include <math.h>

typedef __attribute__((ext_vector_type(16))) __bf16 v16bf;
typedef __attribute__((ext_vector_type(8)))  __bf16 v8bf;
typedef __attribute__((ext_vector_type(8)))  float  v8f;

#define N_ 2048
#define M_ 2048
#define D_ 1024
#define EMB_ 64
#define HEADS_ 16
static constexpr size_t NM_ = (size_t)N_ * (size_t)M_;

// ---------------------------------------------------------------- f32 -> bf16
__global__ void cvt_bf16_kernel(const float* __restrict__ src,
                                __bf16* __restrict__ dst, int n) {
  int i = (blockIdx.x * blockDim.x + threadIdx.x) * 4;
  if (i + 3 < n) {
    float4 v = *(const float4*)(src + i);
    dst[i + 0] = (__bf16)v.x; dst[i + 1] = (__bf16)v.y;
    dst[i + 2] = (__bf16)v.z; dst[i + 3] = (__bf16)v.w;
  } else {
    for (; i < n; ++i) dst[i] = (__bf16)src[i];
  }
}

// Load one 16x32 bf16 WMMA operand tile from a row-major [rows x ld] matrix.
// lane&15 -> row (A) / col (B); half-wave selects K-half per ISA 7.12.2.
__device__ __forceinline__ v16bf load_ab_tile(const __bf16* __restrict__ base,
                                              int row0, int k0, int ld, int lane) {
  int r  = row0 + (lane & 15);
  int kb = k0 + ((lane & 16) ? 8 : 0);
  const __bf16* p = base + (size_t)r * ld + kb;
  v8bf lo = *(const v8bf*)(p);        // K = kb..kb+7      (global_load_b128)
  v8bf hi = *(const v8bf*)(p + 16);   // K = kb+16..kb+23  (global_load_b128)
  v16bf v;
#pragma unroll
  for (int i = 0; i < 8; ++i) { v[i] = lo[i]; v[8 + i] = hi[i]; }
  return v;
}

// ------------------------------------------- q/k: out = bf16(A @ W^T + bias)
// A: [rows x 1024] bf16, W: [1024 x 1024] bf16 (row d, col c) -> symmetric load.
// One wave -> 32x32 output tile (4 wmma accumulators, K looped by 32).
__global__ void gemm_qk_kernel(const __bf16* __restrict__ A,
                               const __bf16* __restrict__ W,
                               const float* __restrict__ bias,
                               __bf16* __restrict__ out, int rows) {
  int wave = (blockIdx.x * blockDim.x + threadIdx.x) >> 5;
  int lane = threadIdx.x & 31;
  int tilesC = D_ / 32;
  int tr = wave / tilesC, tc = wave % tilesC;
  int row0 = tr * 32, col0 = tc * 32;
  if (row0 >= rows) return;
  v8f acc00 = {}, acc01 = {}, acc10 = {}, acc11 = {};
  for (int k0 = 0; k0 < D_; k0 += 32) {
    v16bf a0 = load_ab_tile(A, row0,      k0, D_, lane);
    v16bf a1 = load_ab_tile(A, row0 + 16, k0, D_, lane);
    v16bf b0 = load_ab_tile(W, col0,      k0, D_, lane);
    v16bf b1 = load_ab_tile(W, col0 + 16, k0, D_, lane);
    acc00 = __builtin_amdgcn_wmma_f32_16x16x32_bf16(false, a0, false, b0, (short)0, acc00, false, false);
    acc01 = __builtin_amdgcn_wmma_f32_16x16x32_bf16(false, a0, false, b1, (short)0, acc01, false, false);
    acc10 = __builtin_amdgcn_wmma_f32_16x16x32_bf16(false, a1, false, b0, (short)0, acc10, false, false);
    acc11 = __builtin_amdgcn_wmma_f32_16x16x32_bf16(false, a1, false, b1, (short)0, acc11, false, false);
  }
  int n = lane & 15;
  int hi = (lane & 16) ? 8 : 0;
  v8f* accs[2][2] = {{&acc00, &acc01}, {&acc10, &acc11}};
#pragma unroll
  for (int bi = 0; bi < 2; ++bi)
#pragma unroll
    for (int bj = 0; bj < 2; ++bj) {
      int col = col0 + bj * 16 + n;
      float bv = bias[col];
#pragma unroll
      for (int r = 0; r < 8; ++r) {
        int row = row0 + bi * 16 + hi + r;
        out[(size_t)row * D_ + col] = (__bf16)((*accs[bi][bj])[r] + bv);
      }
    }
}

// ------------------------------------------------- aff = (q @ k^T) * 1/32
__global__ void gemm_aff_kernel(const __bf16* __restrict__ Q,
                                const __bf16* __restrict__ K,
                                float* __restrict__ aff) {
  int wave = (blockIdx.x * blockDim.x + threadIdx.x) >> 5;
  int lane = threadIdx.x & 31;
  int tilesC = M_ / 32;
  int tr = wave / tilesC, tc = wave % tilesC;
  int row0 = tr * 32, col0 = tc * 32;
  if (row0 >= N_) return;
  v8f acc00 = {}, acc01 = {}, acc10 = {}, acc11 = {};
  for (int k0 = 0; k0 < D_; k0 += 32) {
    v16bf a0 = load_ab_tile(Q, row0,      k0, D_, lane);
    v16bf a1 = load_ab_tile(Q, row0 + 16, k0, D_, lane);
    v16bf b0 = load_ab_tile(K, col0,      k0, D_, lane);
    v16bf b1 = load_ab_tile(K, col0 + 16, k0, D_, lane);
    acc00 = __builtin_amdgcn_wmma_f32_16x16x32_bf16(false, a0, false, b0, (short)0, acc00, false, false);
    acc01 = __builtin_amdgcn_wmma_f32_16x16x32_bf16(false, a0, false, b1, (short)0, acc01, false, false);
    acc10 = __builtin_amdgcn_wmma_f32_16x16x32_bf16(false, a1, false, b0, (short)0, acc10, false, false);
    acc11 = __builtin_amdgcn_wmma_f32_16x16x32_bf16(false, a1, false, b1, (short)0, acc11, false, false);
  }
  const float scale = 0.03125f; // 1/sqrt(1024)
  int n = lane & 15;
  int hi = (lane & 16) ? 8 : 0;
  v8f* accs[2][2] = {{&acc00, &acc01}, {&acc10, &acc11}};
#pragma unroll
  for (int bi = 0; bi < 2; ++bi)
#pragma unroll
    for (int bj = 0; bj < 2; ++bj)
#pragma unroll
      for (int r = 0; r < 8; ++r)
        aff[(size_t)(row0 + bi * 16 + hi + r) * M_ + col0 + bj * 16 + n] =
            (*accs[bi][bj])[r] * scale;
}

// ------------- fused single-pass PE stream: pos einsum on WMMA + gated sigmoid
// pos[h,nm] = sum_e Wg[h,e]*PE[e,nm]  ->  A = Wg (16x64, 2 K-chunks), B = PE tile.
// adj = ( sum_h (w+1e-6)/((w+1e-6)+exp(-aff)) > 8 ), w = relu(pos + bias_h).
__global__ void fused_pos_adj_kernel(const float* __restrict__ PE,
                                     const __bf16* __restrict__ Wg,
                                     const float* __restrict__ Wgb,
                                     const float* __restrict__ aff,
                                     float* __restrict__ out) {
  const int TPW = 16; // 16-position tiles per wave
  int lane = threadIdx.x & 31;
  int wave = (blockIdx.x * blockDim.x + threadIdx.x) >> 5;

  // A operands (Wg, bf16) built once per wave
  int h  = lane & 15;
  int kb = (lane & 16) ? 8 : 0;
  const __bf16* wp = Wg + h * EMB_;
  v16bf a0, a1;
#pragma unroll
  for (int i = 0; i < 8; ++i) {
    a0[i]     = wp[kb + i];           a0[8 + i] = wp[kb + 16 + i];
    a1[i]     = wp[32 + kb + i];      a1[8 + i] = wp[32 + kb + 16 + i];
  }
  int hi8 = (lane & 16) ? 8 : 0;
  float br[8];
#pragma unroll
  for (int r = 0; r < 8; ++r) br[r] = Wgb[r + hi8];

  size_t nmBase = (size_t)wave * TPW * 16;
  int col = lane & 15;
  for (int t = 0; t < TPW; ++t) {
    size_t nm0 = nmBase + (size_t)t * 16;
    const float* pb = PE + nm0 + col + (size_t)kb * NM_;
    v16bf b0, b1;
#pragma unroll
    for (int i = 0; i < 8; ++i) {
      b0[i]     = (__bf16)__builtin_nontemporal_load(pb + (size_t)(i)      * NM_);
      b0[8 + i] = (__bf16)__builtin_nontemporal_load(pb + (size_t)(16 + i) * NM_);
      b1[i]     = (__bf16)__builtin_nontemporal_load(pb + (size_t)(32 + i) * NM_);
      b1[8 + i] = (__bf16)__builtin_nontemporal_load(pb + (size_t)(48 + i) * NM_);
    }
    v8f acc = {};
    acc = __builtin_amdgcn_wmma_f32_16x16x32_bf16(false, a0, false, b0, (short)0, acc, false, false);
    acc = __builtin_amdgcn_wmma_f32_16x16x32_bf16(false, a1, false, b1, (short)0, acc, false, false);

    float E = __expf(-aff[nm0 + col]);   // one exp per output position
    float s = 0.f;
#pragma unroll
    for (int r = 0; r < 8; ++r) {
      float w = fmaxf(acc[r] + br[r], 0.f) + 1e-6f;
      s += w * __builtin_amdgcn_rcpf(w + E);   // sigmoid(log(w)+aff)
    }
    s += __shfl_xor(s, 16, 32);               // fold 16 heads across half-waves
    if (lane < 16) out[nm0 + col] = (s > 8.0f) ? 1.0f : 0.0f;
  }
}

// ---------------------------------------------------------------------------
extern "C" void kernel_launch(void* const* d_in, const int* in_sizes, int n_in,
                              void* d_out, int out_size, void* d_ws, size_t ws_size,
                              hipStream_t stream) {
  const float* ref = (const float*)d_in[0];
  const float* sup = (const float*)d_in[1];
  const float* pe  = (const float*)d_in[2];
  const float* wg  = (const float*)d_in[3];
  const float* wgb = (const float*)d_in[4];
  const float* wq  = (const float*)d_in[5];
  const float* wqb = (const float*)d_in[6];
  const float* wk  = (const float*)d_in[7];
  const float* wkb = (const float*)d_in[8];
  float* out = (float*)d_out;

  char* ws = (char*)d_ws;
  const size_t MB = 1u << 20;
  __bf16* qbf   = (__bf16*)(ws + 0 * MB);   // 4 MB
  __bf16* kbf   = (__bf16*)(ws + 4 * MB);   // 4 MB
  __bf16* refbf = (__bf16*)(ws + 8 * MB);   // 4 MB
  __bf16* supbf = (__bf16*)(ws + 12 * MB);  // 4 MB
  __bf16* wqbf  = (__bf16*)(ws + 16 * MB);  // 2 MB
  __bf16* wkbf  = (__bf16*)(ws + 18 * MB);  // 2 MB
  __bf16* wgbf  = (__bf16*)(ws + 20 * MB);  // 2 KB
  float*  aff   = (float*)(ws + 21 * MB);   // 16 MB

  auto cvt = [&](const float* s, __bf16* d, int n) {
    int blocks = (n / 4 + 255) / 256;
    cvt_bf16_kernel<<<blocks, 256, 0, stream>>>(s, d, n);
  };
  cvt(ref, refbf, N_ * D_);
  cvt(sup, supbf, M_ * D_);
  cvt(wq,  wqbf,  D_ * D_);
  cvt(wk,  wkbf,  D_ * D_);
  cvt(wg,  wgbf,  HEADS_ * EMB_);

  {
    int waves = (N_ / 32) * (D_ / 32);        // 2048
    gemm_qk_kernel<<<waves * 32 / 256, 256, 0, stream>>>(refbf, wqbf, wqb, qbf, N_);
    gemm_qk_kernel<<<waves * 32 / 256, 256, 0, stream>>>(supbf, wkbf, wkb, kbf, M_);
  }
  {
    int waves = (N_ / 32) * (M_ / 32);        // 4096
    gemm_aff_kernel<<<waves * 32 / 256, 256, 0, stream>>>(qbf, kbf, aff);
  }
  {
    int waves = (int)(NM_ / 16 / 16);         // 16384
    fused_pos_adj_kernel<<<waves * 32 / 256, 256, 0, stream>>>(pe, wgbf, wgb, aff, out);
  }
}
